// QuantizedPermutationNoiseLinear_6459630813325
// MI455X (gfx1250) — 8-bit matrix operand order repaired and checked by reading, compiled, not run on hardware
//
#include <hip/hip_runtime.h>
#include <hip/hip_bf16.h>

typedef int   v8i __attribute__((ext_vector_type(8)));
typedef float v8f __attribute__((ext_vector_type(8)));
typedef float v2f __attribute__((ext_vector_type(2)));
typedef float v4fv __attribute__((ext_vector_type(4)));
typedef int   v4i_vs __attribute__((vector_size(16)));

#define IN_F   4096
#define OUT_F  1024
#define BS_TOT 8192
#define NROUND 12
#define NGRP   2048
#define KC     32
#define BPAD   36

__device__ __forceinline__ unsigned f32_enc(float f) {
    unsigned u = __float_as_uint(f);
    return (u & 0x80000000u) ? ~u : (u | 0x80000000u);
}
__device__ __forceinline__ float f32_dec(unsigned u) {
    unsigned b = (u & 0x80000000u) ? (u ^ 0x80000000u) : ~u;
    return __uint_as_float(b);
}

__device__ __forceinline__ void cp_async_b128(float* ldst, const float* gsrc) {
#if __has_builtin(__builtin_amdgcn_global_load_async_to_lds_b128)
    __builtin_amdgcn_global_load_async_to_lds_b128(
        (__attribute__((address_space(1))) v4i_vs*)gsrc,
        (__attribute__((address_space(3))) v4i_vs*)ldst, 0, 0);
#else
    *(v4fv*)ldst = *(const v4fv*)gsrc;
#endif
}
__device__ __forceinline__ void wait_async0() {
#if __has_builtin(__builtin_amdgcn_s_wait_asynccnt)
    __builtin_amdgcn_s_wait_asynccnt(0);
#else
    asm volatile("s_wait_asynccnt 0x0" ::: "memory");
#endif
}

__global__ void k_init(unsigned* minmax) {
    if (threadIdx.x < 2) {
        minmax[2 * threadIdx.x]     = 0xFFFFFFFFu;
        minmax[2 * threadIdx.x + 1] = 0u;
    }
}

__global__ void k_wplus_t(const float* __restrict__ w, const float* __restrict__ wn,
                          float* __restrict__ pwT) {
    int i = blockIdx.x * blockDim.x + threadIdx.x;
    int k = i >> 10;
    int n = i & (OUT_F - 1);
    pwT[i] = w[(size_t)n * IN_F + k] + wn[(size_t)n * IN_F + k];
}

__global__ void k_perm_round_w(const float* __restrict__ src, float* __restrict__ dst,
                               const int* __restrict__ perm, const float* __restrict__ mat) {
    int g = blockIdx.x;
    float m00 = mat[4 * g + 0], m01 = mat[4 * g + 1];
    float m10 = mat[4 * g + 2], m11 = mat[4 * g + 3];
    float det = m00 * m11 - m01 * m10;
    float a00 =  m11 / det, a01 = -m10 / det;
    float a10 = -m01 / det, a11 =  m00 / det;
    int p0 = perm[2 * g], p1 = perm[2 * g + 1];
    const float* r0 = src + (size_t)p0 * OUT_F;
    const float* r1 = src + (size_t)p1 * OUT_F;
    float* d0 = dst + (size_t)(2 * g) * OUT_F;
    float* d1 = dst + (size_t)(2 * g + 1) * OUT_F;
    for (int c = threadIdx.x; c < OUT_F; c += blockDim.x) {
        float v0 = r0[c], v1 = r1[c];
        d0[c] = a00 * v0 + a01 * v1;
        d1[c] = a10 * v0 + a11 * v1;
    }
}

__global__ void k_minmax_w(const float* __restrict__ pwT, unsigned* __restrict__ minmax) {
    __shared__ float smn[256], smx[256];
    float mn = 3.4e38f, mx = -3.4e38f;
    int total = IN_F * OUT_F;
    for (int i = blockIdx.x * blockDim.x + threadIdx.x; i < total; i += gridDim.x * blockDim.x) {
        float v = pwT[i];
        mn = fminf(mn, v); mx = fmaxf(mx, v);
    }
    smn[threadIdx.x] = mn; smx[threadIdx.x] = mx;
    __syncthreads();
    for (int s = 128; s > 0; s >>= 1) {
        if (threadIdx.x < s) {
            smn[threadIdx.x] = fminf(smn[threadIdx.x], smn[threadIdx.x + s]);
            smx[threadIdx.x] = fmaxf(smx[threadIdx.x], smx[threadIdx.x + s]);
        }
        __syncthreads();
    }
    if (threadIdx.x == 0) {
        atomicMin(&minmax[2], f32_enc(smn[0]));
        atomicMax(&minmax[3], f32_enc(smx[0]));
    }
}

__global__ void k_scale_zp(const unsigned* __restrict__ minmax, float* __restrict__ params) {
    if (threadIdx.x == 0 && blockIdx.x == 0) {
        for (int t = 0; t < 2; ++t) {
            float lo = f32_dec(minmax[2 * t]);
            float hi = f32_dec(minmax[2 * t + 1]);
            float scale = (hi - lo) / 255.0f;
            float zp = roundf(-128.0f - lo / scale);
            params[2 * t]     = scale;
            params[2 * t + 1] = zp;
        }
    }
}

__global__ void k_quant_w(const float* __restrict__ pwT, const float* __restrict__ params,
                          signed char* __restrict__ qw, int* __restrict__ rsw) {
    __shared__ int ssum[256];
    int n = blockIdx.x;
    float sw = params[2], zw = params[3];
    int sum = 0;
    for (int k = threadIdx.x; k < IN_F; k += blockDim.x) {
        float t = pwT[(size_t)k * OUT_F + n];
        float q = fminf(fmaxf(roundf(t / sw + zw), -128.0f), 127.0f);
        int qi = (int)q;
        qw[(size_t)n * IN_F + k] = (signed char)qi;
        sum += qi;
    }
    ssum[threadIdx.x] = sum;
    __syncthreads();
    for (int s = 128; s > 0; s >>= 1) {
        if (threadIdx.x < s) ssum[threadIdx.x] += ssum[threadIdx.x + s];
        __syncthreads();
    }
    if (threadIdx.x == 0) rsw[n] = ssum[0];
}

__global__ void __launch_bounds__(256)
k_x_chain(const float* __restrict__ x, const float* __restrict__ nx,
          const int* __restrict__ perms, const float* __restrict__ mats,
          unsigned* __restrict__ minmax, const float* __restrict__ params,
          signed char* __restrict__ qx, int* __restrict__ rsx, int mode) {
    __shared__ float s0[IN_F];
    __shared__ float s1[IN_F];
    __shared__ float smn[256], smx[256];
    __shared__ int   ssum[256];
    int row = blockIdx.x;
    const float* xr = x  + (size_t)row * IN_F;
    const float* nr = nx + (size_t)row * IN_F;
    for (int i = threadIdx.x; i < IN_F; i += blockDim.x)
        s0[i] = xr[i] + nr[i];
    __syncthreads();
    float* cur = s0; float* nxt = s1;
    for (int r = 0; r < NROUND; ++r) {
        const int*   pr = perms + r * IN_F;
        const float* mr = mats + (size_t)r * NGRP * 4;
        for (int g = threadIdx.x; g < NGRP; g += blockDim.x) {
            int p0 = pr[2 * g], p1 = pr[2 * g + 1];
            float m00 = mr[4 * g + 0], m01 = mr[4 * g + 1];
            float m10 = mr[4 * g + 2], m11 = mr[4 * g + 3];
            float v0 = cur[p0], v1 = cur[p1];
            nxt[2 * g]     = m00 * v0 + m01 * v1;
            nxt[2 * g + 1] = m10 * v0 + m11 * v1;
        }
        __syncthreads();
        float* t = cur; cur = nxt; nxt = t;
    }
    if (mode == 0) {
        float mn = 3.4e38f, mx = -3.4e38f;
        for (int i = threadIdx.x; i < IN_F; i += blockDim.x) {
            float v = cur[i];
            mn = fminf(mn, v); mx = fmaxf(mx, v);
        }
        smn[threadIdx.x] = mn; smx[threadIdx.x] = mx;
        __syncthreads();
        for (int s = 128; s > 0; s >>= 1) {
            if (threadIdx.x < s) {
                smn[threadIdx.x] = fminf(smn[threadIdx.x], smn[threadIdx.x + s]);
                smx[threadIdx.x] = fmaxf(smx[threadIdx.x], smx[threadIdx.x + s]);
            }
            __syncthreads();
        }
        if (threadIdx.x == 0) {
            atomicMin(&minmax[0], f32_enc(smn[0]));
            atomicMax(&minmax[1], f32_enc(smx[0]));
        }
    } else {
        float sx = params[0], zx = params[1];
        int sum = 0;
        for (int i = threadIdx.x; i < IN_F; i += blockDim.x) {
            float q = fminf(fmaxf(roundf(cur[i] / sx + zx), -128.0f), 127.0f);
            int qi = (int)q;
            qx[(size_t)row * IN_F + i] = (signed char)qi;
            sum += qi;
        }
        ssum[threadIdx.x] = sum;
        __syncthreads();
        for (int s = 128; s > 0; s >>= 1) {
            if (threadIdx.x < s) ssum[threadIdx.x] += ssum[threadIdx.x + s];
            __syncthreads();
        }
        if (threadIdx.x == 0) rsx[row] = ssum[0];
    }
}

__global__ void __launch_bounds__(256)
k_gemm(const signed char* __restrict__ qx, const signed char* __restrict__ qw,
       const int* __restrict__ rsx, const int* __restrict__ rsw,
       const float* __restrict__ params, const float* __restrict__ bias,
       const float* __restrict__ x, const float* __restrict__ nx,
       const float* __restrict__ w, const float* __restrict__ wn,
       float* __restrict__ out) {
    __shared__ __align__(16) float sB[2][64][BPAD];

    int wave = threadIdx.x >> 5;
    int lane = threadIdx.x & 31;
    int rowl = lane & 15;
    int half = lane >> 4;
    int n0 = blockIdx.x * 64;
    int m0 = (blockIdx.y * 8 + wave) * 16;

    v8i acc_i[4];
    #pragma unroll
    for (int t = 0; t < 4; ++t)
        acc_i[t] = (v8i){0, 0, 0, 0, 0, 0, 0, 0};

    const signed char* arow = qx + (size_t)(m0 + rowl) * IN_F + (half ? 8 : 0);
    #pragma unroll 1
    for (int k0 = 0; k0 < IN_F; k0 += 64) {
        v8i a;
        #pragma unroll
        for (int i = 0; i < 8; ++i) {
            int off = (i & 1) * 4 + ((i >> 1) & 1) * 16 + (i >> 2) * 32;
            a[i] = *(const int*)(arow + k0 + off);
        }
        #pragma unroll
        for (int t = 0; t < 4; ++t) {
            const signed char* brow = qw + (size_t)(n0 + t * 16 + rowl) * IN_F + k0 + (half ? 8 : 0);
            v8i b;
            #pragma unroll
            for (int v = 0; v < 8; ++v) {
                int off = (v & 1) * 4 + ((v >> 1) & 1) * 16 + (v >> 2) * 32;
                b[v] = *(const int*)(brow + off);
            }
            acc_i[t] = __builtin_amdgcn_wmma_i32_16x16x64_iu8(true, a, true, b,
                                                              acc_i[t], false, false);
        }
    }

    v8f acc_c[4];
    #pragma unroll
    for (int t = 0; t < 4; ++t)
        acc_c[t] = (v8f){0.f, 0.f, 0.f, 0.f, 0.f, 0.f, 0.f, 0.f};

    const float* xrow = x  + (size_t)(m0 + rowl) * IN_F;
    const float* nrow = nx + (size_t)(m0 + rowl) * IN_F;
    int kh = half * 2;
    int cr = threadIdx.x >> 2;
    int cc = (threadIdx.x & 3) * 8;

    #pragma unroll 1
    for (int kb = 0; kb < IN_F; kb += KC) {
        const float* gwn = wn + (size_t)(n0 + cr) * IN_F + kb + cc;
        const float* gw  = w  + (size_t)(n0 + cr) * IN_F + kb + cc;
        __syncthreads();
        cp_async_b128(&sB[0][cr][cc],     gwn);
        cp_async_b128(&sB[0][cr][cc + 4], gwn + 4);
        cp_async_b128(&sB[1][cr][cc],     gw);
        cp_async_b128(&sB[1][cr][cc + 4], gw + 4);
        wait_async0();
        __syncthreads();
        #pragma unroll
        for (int kk = 0; kk < KC; kk += 4) {
            int k = kb + kk;
            v2f a2; a2.x = nrow[k + kh]; a2.y = nrow[k + kh + 1];
            v2f a1; a1.x = xrow[k + kh] + a2.x; a1.y = xrow[k + kh + 1] + a2.y;
            #pragma unroll
            for (int t = 0; t < 4; ++t) {
                const float* b1p = &sB[0][t * 16 + rowl][kk + kh];
                const float* b2p = &sB[1][t * 16 + rowl][kk + kh];
                v2f b1; b1.x = b1p[0]; b1.y = b1p[1];
                v2f b2; b2.x = b2p[0]; b2.y = b2p[1];
                acc_c[t] = __builtin_amdgcn_wmma_f32_16x16x4_f32(false, a1, false, b1,
                                                                 (short)0, acc_c[t], false, false);
                acc_c[t] = __builtin_amdgcn_wmma_f32_16x16x4_f32(false, a2, false, b2,
                                                                 (short)0, acc_c[t], false, false);
            }
        }
    }

    float sx = params[0], zx = params[1], sw = params[2], zw = params[3];
    double ssd = (double)sx * (double)sw;
    int rsx_l[8];
    #pragma unroll
    for (int r = 0; r < 8; ++r) rsx_l[r] = rsx[m0 + r + half * 8];
    #pragma unroll
    for (int t = 0; t < 4; ++t) {
        int n = n0 + t * 16 + rowl;
        float bn = bias[n];
        int rswn = rsw[n];
        #pragma unroll
        for (int r = 0; r < 8; ++r) {
            int m = m0 + r + half * 8;
            double dq = (double)acc_i[t][r]
                      - (double)zw * (double)rsx_l[r]
                      - (double)zx * (double)rswn
                      + (double)IN_F * (double)zx * (double)zw;
            out[(size_t)m * OUT_F + n] = (float)(ssd * dq) + bn - acc_c[t][r];
        }
    }
}

extern "C" void kernel_launch(void* const* d_in, const int* in_sizes, int n_in,
                              void* d_out, int out_size, void* d_ws, size_t ws_size,
                              hipStream_t stream) {
    const float* x     = (const float*)d_in[0];
    const float* wgt   = (const float*)d_in[1];
    const float* bias  = (const float*)d_in[2];
    const float* wn    = (const float*)d_in[3];
    const float* nx    = (const float*)d_in[4];
    const int*   perms = (const int*)d_in[5];
    const float* mats  = (const float*)d_in[6];
    float* out = (float*)d_out;

    char* ws = (char*)d_ws;
    unsigned* minmax = (unsigned*)ws;
    float*    params = (float*)(ws + 16);
    int*      rsx    = (int*)(ws + 32);
    int*      rsw    = (int*)(ws + 32 + 32768);
    float*    pwA    = (float*)(ws + 65536);
    float*    pwB    = pwA + (size_t)IN_F * OUT_F;
    signed char* qx  = (signed char*)(pwB + (size_t)IN_F * OUT_F);
    signed char* qw  = qx + (size_t)BS_TOT * IN_F;

    k_init<<<1, 32, 0, stream>>>(minmax);
    k_wplus_t<<<(IN_F * OUT_F) / 256, 256, 0, stream>>>(wgt, wn, pwA);

    for (int r = 0; r < NROUND; ++r) {
        const float* src = (r & 1) ? pwB : pwA;
        float*       dst = (r & 1) ? pwA : pwB;
        k_perm_round_w<<<NGRP, 256, 0, stream>>>(src, dst, perms + r * IN_F,
                                                 mats + (size_t)r * NGRP * 4);
    }
    k_minmax_w<<<1024, 256, 0, stream>>>(pwA, minmax);
    k_x_chain<<<BS_TOT, 256, 0, stream>>>(x, nx, perms, mats, minmax, params, qx, rsx, 0);
    k_scale_zp<<<1, 32, 0, stream>>>(minmax, params);
    k_quant_w<<<OUT_F, 256, 0, stream>>>(pwA, params, qw, rsw);
    k_x_chain<<<BS_TOT, 256, 0, stream>>>(x, nx, perms, mats, minmax, params, qx, rsx, 1);

    dim3 grid(OUT_F / 64, BS_TOT / 128);
    k_gemm<<<grid, 256, 0, stream>>>(qx, qw, rsx, rsw, params, bias,
                                     x, nx, wgt, wn, out);
}
